// ContentPhasorStream_37082747633852
// MI455X (gfx1250) — matrix operand order repaired and checked by reading, not compiled, not run on hardware
//
#include <hip/hip_runtime.h>
#include <math.h>

// ---------------------------------------------------------------------------
// ContentPhasorStream on MI455X (gfx1250, wave32, WMMA + TDM)
//
// retrieved[b,l,d] = sum_{t<=l} ( f_q(l) . f_k(t) ) * v[t,d] / sqrt((l+1)*K)
// with f(phi) = [cos(phi), sin(phi)] in R^{64}.  Implemented as chunked
// (flash-style) linear attention: all heavy math in v_wmma_f32_16x16x32_bf16,
// V-chunk staging via the Tensor Data Mover (tensor_load_to_lds).
// ---------------------------------------------------------------------------

#define B_   2
#define L_   2048
#define D_   256
#define KP_  32          // phase dim K
#define F_   64          // feature dim = 2*K
#define C_   128         // chunk length
#define NC_  (L_ / C_)   // chunks per batch = 16
#define PI_F 3.14159265358979323846f

typedef __bf16 bf16;
typedef __bf16 bf16x16 __attribute__((ext_vector_type(16)));
typedef float  f32x8   __attribute__((ext_vector_type(8)));
typedef unsigned int u32x4 __attribute__((ext_vector_type(4)));
typedef int  i32x8 __attribute__((ext_vector_type(8)));
typedef int  i32x4 __attribute__((ext_vector_type(4)));

// ---------------------------------------------------------------------------
// WMMA helpers (contiguous k order: lane half h holds k = 16h + j, wave32)
// ---------------------------------------------------------------------------

__device__ inline f32x8 wmma_bf16(bf16x16 a, bf16x16 b, f32x8 c) {
  return __builtin_amdgcn_wmma_f32_16x16x32_bf16(false, a, false, b,
                                                 (short)0, c, false, false);
}

// A fragment (16x32) from row-major [M x K] bf16, leading dim ld.
__device__ inline bf16x16 ldA(const bf16* p, int ld, int m0, int k0, int lane) {
  const int r  = m0 + (lane & 15);
  const int hi = (lane >> 4) * 16;
  bf16x16 a;
#pragma unroll
  for (int j = 0; j < 16; ++j) {
    const int k = k0 + j + hi;
    a[j] = p[r * ld + k];
  }
  return a;
}

// A fragment where logical A[m][k] = p[k*ld + m]  (i.e. A = p^T)
__device__ inline bf16x16 ldAT(const bf16* p, int ld, int m0, int k0, int lane) {
  const int m  = m0 + (lane & 15);
  const int hi = (lane >> 4) * 16;
  bf16x16 a;
#pragma unroll
  for (int j = 0; j < 16; ++j) {
    const int k = k0 + j + hi;
    a[j] = p[k * ld + m];
  }
  return a;
}

// B fragment (32x16) from row-major [K x N] bf16.
__device__ inline bf16x16 ldB(const bf16* p, int ld, int k0, int n0, int lane) {
  const int n  = n0 + (lane & 15);
  const int kk = k0 + (lane >> 4) * 16;
  bf16x16 b;
#pragma unroll
  for (int j = 0; j < 16; ++j) b[j] = p[(kk + j) * ld + n];
  return b;
}

// B fragment where logical B[k][n] = p[n*ld + k] (B = p^T), bf16 source.
// 16 consecutive bf16 per lane -> wide coalesced loads, no cvt in hot loop.
__device__ inline bf16x16 ldBT(const bf16* p, int ld, int k0, int n0, int lane) {
  const int n  = n0 + (lane & 15);
  const int kk = k0 + (lane >> 4) * 16;
  const bf16* row = p + n * ld + kk;
  bf16x16 b;
#pragma unroll
  for (int j = 0; j < 16; ++j) b[j] = row[j];
  return b;
}

// B fragment from row-major [K x N] fp32, converted to bf16 on load.
__device__ inline bf16x16 ldBf32(const float* p, int ld, int k0, int n0, int lane) {
  const int n  = n0 + (lane & 15);
  const int kk = k0 + (lane >> 4) * 16;
  bf16x16 b;
#pragma unroll
  for (int j = 0; j < 16; ++j) b[j] = (bf16)p[(kk + j) * ld + n];
  return b;
}

// Branch-free tanh: tanh(x) = (e^{2x}-1) * rcp(e^{2x}+1).  Input clamped;
// emits v_exp_f32 + v_rcp_f32 instead of libm's exec-mask branches.
__device__ inline float fast_tanh(float x) {
  x = fminf(fmaxf(x, -10.0f), 10.0f);
  const float e = __expf(2.0f * x);
  return (e - 1.0f) * __builtin_amdgcn_rcpf(e + 1.0f);
}

// ---------------------------------------------------------------------------
// Kernel 0: convert all five weight matrices fp32 -> bf16 once.
// Segment layout in wbuf: kw1[65536] qw1[65536] vw[65536] kw2[8192] qw2[8192]
// ---------------------------------------------------------------------------
#define WSEG1 (D_ * D_)          // 65536
#define WSEG2 (KP_ * D_)         // 8192
#define WTOT  (3 * WSEG1 + 2 * WSEG2)

__global__ __launch_bounds__(256) void cvt_weights_kernel(
    const float* __restrict__ kw1, const float* __restrict__ qw1,
    const float* __restrict__ vw,  const float* __restrict__ kw2,
    const float* __restrict__ qw2, bf16* __restrict__ o)
{
  const int i = blockIdx.x * 256 + threadIdx.x;
  if (i >= WTOT) return;
  const float* src;
  int off;
  if      (i < WSEG1)           { src = kw1; off = i; }
  else if (i < 2 * WSEG1)       { src = qw1; off = i - WSEG1; }
  else if (i < 3 * WSEG1)       { src = vw;  off = i - 2 * WSEG1; }
  else if (i < 3 * WSEG1 + WSEG2) { src = kw2; off = i - 3 * WSEG1; }
  else                          { src = qw2; off = i - 3 * WSEG1 - WSEG2; }
  o[i] = (bf16)src[off];
}

// ---------------------------------------------------------------------------
// Kernel 1: fused encoder (bf16 WMMA GEMMs + tanh/sincos epilogues).
// One WG = 32 rows of L, 2 waves (each wave owns 16 rows).
// ---------------------------------------------------------------------------
__global__ __launch_bounds__(64) void encode_kernel(
    const float* __restrict__ x,
    const bf16* __restrict__ wk1, const float* __restrict__ kb1,
    const bf16* __restrict__ wk2, const float* __restrict__ kb2,
    const bf16* __restrict__ wq1, const float* __restrict__ qb1,
    const bf16* __restrict__ wq2, const float* __restrict__ qb2,
    const bf16* __restrict__ wv,  const float* __restrict__ vb,
    bf16* __restrict__ fk, bf16* __restrict__ fq, bf16* __restrict__ vout)
{
  __shared__ bf16 lx[32][D_ + 8];   // x tile (bf16)
  __shared__ bf16 lh[32][D_ + 8];   // hidden tile (bf16)

  const int tid  = threadIdx.x;
  const int lane = tid & 31;
  const int wave = tid >> 5;
  const int wg   = blockIdx.x;               // B * (L/32) = 128
  const int b    = wg / (L_ / 32);
  const int l0   = (wg % (L_ / 32)) * 32;
  const float* xt = x + ((size_t)b * L_ + l0) * D_;

  if (tid == 0) {   // global_prefetch_b8 hints for the weight images
    __builtin_prefetch(wk1, 0, 1);
    __builtin_prefetch(wq1, 0, 1);
    __builtin_prefetch(wv, 0, 1);
  }

  // stage x tile as bf16 into LDS
  for (int i = tid; i < 32 * D_; i += 64) {
    const int r = i >> 8, cc = i & (D_ - 1);
    lx[r][cc] = (bf16)xt[r * D_ + cc];
  }
  __syncthreads();

  const int m0 = wave * 16;   // this wave's 16 rows within the tile

  // pass 0 = key path, pass 1 = query path
  for (int pass = 0; pass < 2; ++pass) {
    const bf16*  w1 = pass ? wq1 : wk1;
    const float* b1 = pass ? qb1 : kb1;
    const bf16*  w2 = pass ? wq2 : wk2;
    const float* b2 = pass ? qb2 : kb2;
    bf16* fo = pass ? fq : fk;

    // h = tanh(x @ w1^T + b1) : 16 rows x 256 cols per wave
    for (int n0 = 0; n0 < D_; n0 += 16) {
      f32x8 acc = {};
      for (int k0 = 0; k0 < D_; k0 += 32) {
        bf16x16 a  = ldA(&lx[0][0], D_ + 8, m0, k0, lane);
        bf16x16 bb = ldBT(w1, D_, k0, n0, lane);
        acc = wmma_bf16(a, bb, acc);
      }
      const int col   = n0 + (lane & 15);
      const int rbase = m0 + ((lane >> 4) << 3);
      const float bias = b1[col];
#pragma unroll
      for (int i = 0; i < 8; ++i)
        lh[rbase + i][col] = (bf16)fast_tanh(acc[i] + bias);
    }
    __syncthreads();

    // phase = tanh(h @ w2^T + b2) * pi  -> cos/sin features
    for (int n0 = 0; n0 < KP_; n0 += 16) {
      f32x8 acc = {};
      for (int k0 = 0; k0 < D_; k0 += 32) {
        bf16x16 a  = ldA(&lh[0][0], D_ + 8, m0, k0, lane);
        bf16x16 bb = ldBT(w2, D_, k0, n0, lane);
        acc = wmma_bf16(a, bb, acc);
      }
      const int col   = n0 + (lane & 15);
      const int rbase = m0 + ((lane >> 4) << 3);
      const float bias = b2[col];
#pragma unroll
      for (int i = 0; i < 8; ++i) {
        const float ph  = fast_tanh(acc[i] + bias) * PI_F;
        const size_t row = (size_t)b * L_ + (l0 + rbase + i);
        fo[row * F_ + col]       = (bf16)__cosf(ph);
        fo[row * F_ + KP_ + col] = (bf16)__sinf(ph);
      }
    }
    __syncthreads();
  }

  // value = x @ vw^T + vb
  for (int n0 = 0; n0 < D_; n0 += 16) {
    f32x8 acc = {};
    for (int k0 = 0; k0 < D_; k0 += 32) {
      bf16x16 a  = ldA(&lx[0][0], D_ + 8, m0, k0, lane);
      bf16x16 bb = ldBT(wv, D_, k0, n0, lane);
      acc = wmma_bf16(a, bb, acc);
    }
    const int col   = n0 + (lane & 15);
    const int rbase = m0 + ((lane >> 4) << 3);
    const float bias = vb[col];
#pragma unroll
    for (int i = 0; i < 8; ++i) {
      const size_t row = (size_t)b * L_ + (l0 + rbase + i);
      vout[row * D_ + col] = (bf16)(acc[i] + bias);
    }
  }
}

// ---------------------------------------------------------------------------
// Kernel 2: per-chunk state  S_c = Fk_c^T @ V_c   (64 x 256, fp32)
// V chunk (128x256 bf16 = 64 KB, contiguous) is staged into LDS by the
// Tensor Data Mover (tensor_load_to_lds + s_wait_tensorcnt).
// ---------------------------------------------------------------------------
__global__ __launch_bounds__(128) void chunk_state_kernel(
    const bf16* __restrict__ fk, const bf16* __restrict__ vbf,
    float* __restrict__ S)
{
  __shared__ bf16 lv[C_ * D_];                  // 64 KB V chunk

  const int tid = threadIdx.x, lane = tid & 31, wave = tid >> 5;
  const int wg = blockIdx.x;                    // B*NC
  const int b = wg / NC_, c = wg % NC_;
  const int t0 = c * C_;
  const bf16* fkc = fk  + ((size_t)b * L_ + t0) * F_;   // C x F
  const bf16* vc  = vbf + ((size_t)b * L_ + t0) * D_;   // C x D (contiguous)
  float* Sc = S + (size_t)(b * NC_ + c) * F_ * D_;

#if __has_builtin(__builtin_amdgcn_tensor_load_to_lds)
  if (wave == 0) {
    // D# for a 1D copy of C_*D_ = 32768 bf16 elements (ISA ch.8).
    const unsigned long long ga = (unsigned long long)(uintptr_t)vc;
    const unsigned int lbase = (unsigned int)(uintptr_t)(void*)lv;
    u32x4 g0;
    g0.x = 1u;                                           // count=1, user D#
    g0.y = lbase;                                        // lds_addr
    g0.z = (unsigned int)ga;                             // global_addr[31:0]
    g0.w = (unsigned int)((ga >> 32) & 0x01FFFFFFull)    // global_addr[56:32]
         | (2u << 30);                                   // type=2 ("image")
    i32x8 g1;
    g1[0] = 0x00010000;        // workgroup_mask=0, data_size=1 (2 bytes)
    g1[1] = (int)0x80000000;   // tensor_dim0 = 32768 (low16 in [31:16])
    g1[2] = 0x00010000;        // tensor_dim0 hi=0, tensor_dim1 = 1
    g1[3] = (int)0x80000000;   // tile_dim0 = 32768
    g1[4] = 0;                 // tile_dim1 = tile_dim2 = 0 (unused)
    g1[5] = 32768;             // tensor_dim0_stride (low 32)
    g1[6] = 0;
    g1[7] = 0;
    i32x4 z4 = {0, 0, 0, 0};
#if defined(__clang_major__) && (__clang_major__ >= 23)
    i32x8 z8 = {0, 0, 0, 0, 0, 0, 0, 0};
    __builtin_amdgcn_tensor_load_to_lds(g0, g1, z4, z4, z8, 0);
#else
    __builtin_amdgcn_tensor_load_to_lds(g0, g1, z4, z4, 0);
#endif
    __builtin_amdgcn_s_wait_tensorcnt(0);
  }
  __syncthreads();
#else
  for (int i = tid; i < C_ * D_; i += 128) lv[i] = vc[i];
  __syncthreads();
#endif

  const int m0 = wave * 16;   // feature rows
  for (int n0 = 0; n0 < D_; n0 += 16) {
    f32x8 acc = {};
    for (int k0 = 0; k0 < C_; k0 += 32) {
      bf16x16 a  = ldAT(fkc, F_, m0, k0, lane);  // A[m=f][k=t] = fkc[t*F + f]
      bf16x16 bb = ldB(lv, D_, k0, n0, lane);    // V from LDS
      acc = wmma_bf16(a, bb, acc);
    }
    const int col   = n0 + (lane & 15);
    const int rbase = m0 + ((lane >> 4) << 3);
#pragma unroll
    for (int i = 0; i < 8; ++i)
      Sc[(rbase + i) * D_ + col] = acc[i];
  }
}

// ---------------------------------------------------------------------------
// Kernel 3: exclusive prefix over the 16 chunk states (per batch), fp32.
// ---------------------------------------------------------------------------
__global__ __launch_bounds__(256) void prefix_kernel(
    const float* __restrict__ S, float* __restrict__ P)
{
  const int b = blockIdx.x;
  const size_t stride = (size_t)F_ * D_;          // 16384 elements per state
  for (int e = threadIdx.x; e < (int)stride; e += 256) {
    float run = 0.f;
    for (int c = 0; c < NC_; ++c) {
      const size_t idx = (size_t)(b * NC_ + c) * stride + e;
      P[idx] = run;
      run += S[idx];
    }
  }
}

// ---------------------------------------------------------------------------
// Kernel 4: per-chunk output
//   out = [ causal(Fq @ Fk^T) @ V  +  Fq @ P_c ] * rsqrt((l+1)*K)
// One WG per (b, chunk); 4 waves, each owns 32 chunk rows.
// ---------------------------------------------------------------------------
__global__ __launch_bounds__(128) void output_kernel(
    const bf16* __restrict__ fq, const bf16* __restrict__ fk,
    const bf16* __restrict__ vbf, const float* __restrict__ P,
    float* __restrict__ out)
{
  __shared__ bf16 lq[C_][F_ + 8];        // Fq chunk
  __shared__ bf16 lsc[C_][C_ + 8];       // masked scores (bf16)

  const int tid = threadIdx.x, lane = tid & 31, wave = tid >> 5;
  const int wg = blockIdx.x;
  const int b = wg / NC_, c = wg % NC_;
  const int t0 = c * C_;
  const bf16* fqc = fq  + ((size_t)b * L_ + t0) * F_;
  const bf16* fkc = fk  + ((size_t)b * L_ + t0) * F_;
  const bf16* vc  = vbf + ((size_t)b * L_ + t0) * D_;
  const float* Pc = P + (size_t)(b * NC_ + c) * F_ * D_;
  float* oc = out + ((size_t)b * L_ + t0) * D_;

  for (int i = tid; i < C_ * F_; i += 128) {
    const int r = i >> 6, f = i & (F_ - 1);
    lq[r][f] = fqc[r * F_ + f];
  }
  __syncthreads();

  const int mrow0 = wave * 32;           // this wave's 32 rows

  // scores = causal_mask(Fq @ Fk^T), inclusive diagonal
  for (int mt = 0; mt < 2; ++mt) {
    const int m0 = mrow0 + mt * 16;
    for (int n0 = 0; n0 < C_; n0 += 16) {
      f32x8 acc = {};
      for (int k0 = 0; k0 < F_; k0 += 32) {
        bf16x16 a  = ldA(&lq[0][0], F_ + 8, m0, k0, lane);
        bf16x16 bb = ldBT(fkc, F_, k0, n0, lane);   // B[k=f][n=t'] = fk[t'][f]
        acc = wmma_bf16(a, bb, acc);
      }
      const int col   = n0 + (lane & 15);
      const int rbase = m0 + ((lane >> 4) << 3);
#pragma unroll
      for (int i = 0; i < 8; ++i) {
        const int r = rbase + i;
        lsc[r][col] = (bf16)((col <= r) ? acc[i] : 0.f);
      }
    }
  }
  __syncthreads();

  // out = scores @ V + Fq @ P, then normalize
  for (int mt = 0; mt < 2; ++mt) {
    const int m0 = mrow0 + mt * 16;
    for (int n0 = 0; n0 < D_; n0 += 16) {
      f32x8 acc = {};
      for (int k0 = 0; k0 < F_; k0 += 32) {          // inter-chunk: Fq @ P
        bf16x16 a  = ldA(&lq[0][0], F_ + 8, m0, k0, lane);
        bf16x16 bb = ldBf32(Pc, D_, k0, n0, lane);
        acc = wmma_bf16(a, bb, acc);
      }
      for (int k0 = 0; k0 < C_; k0 += 32) {          // intra-chunk: scores @ V
        bf16x16 a  = ldA(&lsc[0][0], C_ + 8, m0, k0, lane);
        bf16x16 bb = ldB(vc, D_, k0, n0, lane);
        acc = wmma_bf16(a, bb, acc);
      }
      const int col   = n0 + (lane & 15);
      const int rbase = m0 + ((lane >> 4) << 3);
#pragma unroll
      for (int i = 0; i < 8; ++i) {
        const int r = rbase + i;
        const float norm = rsqrtf((float)(t0 + r + 1) * (float)KP_);
        oc[r * D_ + col] = acc[i] * norm;
      }
    }
  }
}

// ---------------------------------------------------------------------------
// Launcher
// ---------------------------------------------------------------------------
extern "C" void kernel_launch(void* const* d_in, const int* in_sizes, int n_in,
                              void* d_out, int out_size, void* d_ws, size_t ws_size,
                              hipStream_t stream) {
  const float* x   = (const float*)d_in[0];
  const float* kw1 = (const float*)d_in[1];
  const float* kb1 = (const float*)d_in[2];
  const float* kw2 = (const float*)d_in[3];
  const float* kb2 = (const float*)d_in[4];
  const float* qw1 = (const float*)d_in[5];
  const float* qb1 = (const float*)d_in[6];
  const float* qw2 = (const float*)d_in[7];
  const float* qb2 = (const float*)d_in[8];
  const float* vw  = (const float*)d_in[9];
  const float* vb  = (const float*)d_in[10];

  // workspace layout
  char* ws = (char*)d_ws;
  const size_t w_bytes  = (size_t)WTOT * sizeof(bf16);                 // 416 KB
  const size_t fk_bytes = (size_t)B_ * L_ * F_ * sizeof(bf16);         // 512 KB
  const size_t v_bytes  = (size_t)B_ * L_ * D_ * sizeof(bf16);         // 2 MB
  const size_t s_bytes  = (size_t)B_ * NC_ * F_ * D_ * sizeof(float);  // 2 MB
  bf16*  wbuf = (bf16*)(ws);
  bf16*  fk   = (bf16*)(ws + w_bytes);
  bf16*  fq   = (bf16*)(ws + w_bytes + fk_bytes);
  bf16*  vbf  = (bf16*)(ws + w_bytes + 2 * fk_bytes);
  float* S    = (float*)(ws + w_bytes + 2 * fk_bytes + v_bytes);
  float* P    = (float*)(ws + w_bytes + 2 * fk_bytes + v_bytes + s_bytes);

  bf16* wk1 = wbuf;
  bf16* wq1 = wbuf + WSEG1;
  bf16* wv  = wbuf + 2 * WSEG1;
  bf16* wk2 = wbuf + 3 * WSEG1;
  bf16* wq2 = wbuf + 3 * WSEG1 + WSEG2;

  cvt_weights_kernel<<<(WTOT + 255) / 256, 256, 0, stream>>>(
      kw1, qw1, vw, kw2, qw2, wbuf);
  encode_kernel<<<B_ * (L_ / 32), 64, 0, stream>>>(
      x, wk1, kb1, wk2, kb2, wq1, qb1, wq2, qb2, wv, vb, fk, fq, vbf);
  chunk_state_kernel<<<B_ * NC_, 128, 0, stream>>>(fk, vbf, S);
  prefix_kernel<<<B_, 256, 0, stream>>>(S, P);
  output_kernel<<<B_ * NC_, 128, 0, stream>>>(fq, fk, vbf, P, (float*)d_out);
}
